// AttentionModule_4750233830032
// MI455X (gfx1250) — matrix operand order repaired and checked by reading, not compiled, not run on hardware
//
#include <hip/hip_runtime.h>
#include <hip/hip_bf16.h>

typedef __attribute__((ext_vector_type(8)))  _Float16 v8h;
typedef __attribute__((ext_vector_type(16))) _Float16 v16h;
typedef __attribute__((ext_vector_type(8)))  float    v8f;
typedef int vi4 __attribute__((vector_size(16)));

#define WMMA16(a, b, c) \
  __builtin_amdgcn_wmma_f32_16x16x32_f16(false, (a), false, (b), (short)0, (c), false, false)

#if defined(__gfx1250__) && __has_builtin(__builtin_amdgcn_global_load_async_to_lds_b128)
#define HAVE_ASYNC_LDS 1
#else
#define HAVE_ASYNC_LDS 0
#endif

#if HAVE_ASYNC_LDS
// 16-byte global -> LDS async DMA (tracked by ASYNCcnt, not VGPR-returning).
// Probe-confirmed signature: (v4i addrspace(1)*, v4i addrspace(3)*, imm offset, imm cpol)
__device__ __forceinline__ void async_copy16(const void* g, void* l) {
  __builtin_amdgcn_global_load_async_to_lds_b128(
      (__attribute__((address_space(1))) vi4*)(uintptr_t)g,
      (__attribute__((address_space(3))) vi4*)(uint32_t)(uintptr_t)l, 0, 0);
}
#endif

__device__ __forceinline__ void wait_async_zero() {
#if HAVE_ASYNC_LDS
#if __has_builtin(__builtin_amdgcn_s_wait_asynccnt)
  __builtin_amdgcn_s_wait_asynccnt(0);
#else
  asm volatile("s_wait_asynccnt 0x0" ::: "memory");
#endif
#endif
}

// Pack two 8-halfword groups (per-lane) into a 16-wide A/B fragment register set.
__device__ __forceinline__ v16h pack2(const _Float16* lo, const _Float16* hi) {
  v8h a = *(const v8h*)lo;
  v8h b = *(const v8h*)hi;
  v16h r;
#pragma unroll
  for (int i = 0; i < 8; ++i) { r[i] = a[i]; r[i + 8] = b[i]; }
  return r;
}

// ---------------------------------------------------------------------------
// fp32 -> fp16 conversion (weights / Er mirrors), grid-stride
// ---------------------------------------------------------------------------
__global__ void cvt_f32_to_f16(const float* __restrict__ src,
                               _Float16* __restrict__ dst, int n) {
  for (int i = blockIdx.x * blockDim.x + threadIdx.x; i < n;
       i += gridDim.x * blockDim.x)
    dst[i] = (_Float16)src[i];
}

// ---------------------------------------------------------------------------
// h = concat(x, position); h16 mirror; non-pad mask from x[:, :, 0]
// grid = B*L blocks x 256 threads
// ---------------------------------------------------------------------------
__global__ void prep_kernel(const float* __restrict__ x,
                            const float* __restrict__ pos,
                            float* __restrict__ h, _Float16* __restrict__ h16,
                            float* __restrict__ npd) {
  const int DF = 448, DP = 64, D = 512;
  size_t row = blockIdx.x;
  int c = threadIdx.x;
#pragma unroll
  for (int t = 0; t < 2; ++t) {
    int cc = c + t * 256;
    float vv = (cc < DF) ? x[row * DF + cc] : pos[row * DP + (cc - DF)];
    h[row * D + cc]   = vv;
    h16[row * D + cc] = (_Float16)vv;
  }
  if (c == 0) npd[row] = (x[row * DF] != 0.0f) ? 1.0f : 0.0f;
}

// ---------------------------------------------------------------------------
// WMMA GEMM: C[M,N] = A[M,K] (f16, row-major) @ B[K,N] (f16, row-major)
// Software-pipelined, double-buffered LDS; A tile staged via async LDS-DMA
// when available. Epilogue: optional fp32 residual, ReLU, fp32/f16 stores.
// 256 threads = 8 waves as 4(M) x 2(N); block tile 128x128; K-step 32.
// ---------------------------------------------------------------------------
__global__ __launch_bounds__(256) void gemm_f16_kernel(
    const _Float16* __restrict__ A, const _Float16* __restrict__ Bm,
    int M, int N, int K, const float* __restrict__ Res,
    float* __restrict__ Cf, _Float16* __restrict__ Ch, int relu) {
  __shared__ _Float16 As[2][128 * 32];  // row-major 128 x 32
  __shared__ _Float16 Bt[2][128 * 32];  // transposed: [n][k], K-contiguous

  const int tid = threadIdx.x, lane = tid & 31, wid = tid >> 5;
  const int wm = wid >> 1, wn = wid & 1;
  const int bm = blockIdx.y, bn = blockIdx.x;
  const int row16 = lane & 15, hi = lane >> 4, koff = hi << 4;

  v8f acc[2][4] = {};
  const int nsteps = K >> 5;

  // per-thread staging coordinates (two 8-half chunks each for A and B)
  int ar[2], ac[2], bk[2], bc[2];
#pragma unroll
  for (int it = 0; it < 2; ++it) {
    int idx = tid + (it << 8);
    ar[it] = idx >> 2;  ac[it] = (idx & 3) << 3;   // A: 128 rows x 4 chunks
    bk[it] = idx >> 4;  bc[it] = (idx & 15) << 3;  // B: 32 k-rows x 16 chunks
  }

  v8h breg[2];
#if !HAVE_ASYNC_LDS
  v8h areg[2];
#endif

  // ---- prologue: stage K-tile 0 into buffer 0 ----
#pragma unroll
  for (int it = 0; it < 2; ++it) {
#if HAVE_ASYNC_LDS
    async_copy16(A + (size_t)(bm * 128 + ar[it]) * K + ac[it],
                 &As[0][ar[it] * 32 + ac[it]]);
#else
    areg[it] = *(const v8h*)(A + (size_t)(bm * 128 + ar[it]) * K + ac[it]);
#endif
    breg[it] = *(const v8h*)(Bm + (size_t)bk[it] * N + bn * 128 + bc[it]);
  }
#pragma unroll
  for (int it = 0; it < 2; ++it) {
#if !HAVE_ASYNC_LDS
    *(v8h*)&As[0][ar[it] * 32 + ac[it]] = areg[it];
#endif
#pragma unroll
    for (int i = 0; i < 8; ++i) Bt[0][(bc[it] + i) * 32 + bk[it]] = breg[it][i];
  }
  wait_async_zero();
  __syncthreads();

  for (int ks = 0; ks < nsteps; ++ks) {
    const int buf = ks & 1, nbuf = buf ^ 1;
    const bool more = (ks + 1 < nsteps);
    const int k1 = (ks + 1) << 5;

    // ---- issue next tile's global traffic before computing this one ----
    if (more) {
#pragma unroll
      for (int it = 0; it < 2; ++it) {
#if HAVE_ASYNC_LDS
        async_copy16(A + (size_t)(bm * 128 + ar[it]) * K + k1 + ac[it],
                     &As[nbuf][ar[it] * 32 + ac[it]]);
#else
        areg[it] = *(const v8h*)(A + (size_t)(bm * 128 + ar[it]) * K + k1 + ac[it]);
#endif
        breg[it] = *(const v8h*)(Bm + (size_t)(k1 + bk[it]) * N + bn * 128 + bc[it]);
      }
    }

    // ---- compute current tile from LDS ----
    v16h af[2], bf[4];
#pragma unroll
    for (int mt = 0; mt < 2; ++mt) {
      const _Float16* p = &As[buf][(wm * 32 + mt * 16 + row16) * 32 + koff];
      af[mt] = pack2(p, p + 8);   // A layout: K 0-15 (lo half), 16-31 (hi half)
    }
#pragma unroll
    for (int nt = 0; nt < 4; ++nt) {
      const _Float16* p = &Bt[buf][(wn * 64 + nt * 16 + row16) * 32 + (hi << 4)];
      bf[nt] = pack2(p, p + 8);   // B layout: K 0-15 (lo half), 16-31 (hi half)
    }
#pragma unroll
    for (int mt = 0; mt < 2; ++mt)
#pragma unroll
      for (int nt = 0; nt < 4; ++nt)
        acc[mt][nt] = WMMA16(af[mt], bf[nt], acc[mt][nt]);

    // ---- commit next tile into the other buffer ----
    if (more) {
#pragma unroll
      for (int it = 0; it < 2; ++it) {
#if !HAVE_ASYNC_LDS
        *(v8h*)&As[nbuf][ar[it] * 32 + ac[it]] = areg[it];
#endif
#pragma unroll
        for (int i = 0; i < 8; ++i) Bt[nbuf][(bc[it] + i) * 32 + bk[it]] = breg[it][i];
      }
      wait_async_zero();
      __syncthreads();
    }
  }

  // epilogue: C element (r) maps to row r + 8*hi, col = lane&15 of the 16x16 tile
#pragma unroll
  for (int mt = 0; mt < 2; ++mt)
#pragma unroll
    for (int nt = 0; nt < 4; ++nt)
#pragma unroll
      for (int r = 0; r < 8; ++r) {
        int grow = bm * 128 + wm * 32 + mt * 16 + r + (hi << 3);
        int gcol = bn * 128 + wn * 64 + nt * 16 + row16;
        float vv = acc[mt][nt][r];
        if (Res)  vv += Res[(size_t)grow * N + gcol];
        if (relu) vv = fmaxf(vv, 0.0f);
        if (Cf)   Cf[(size_t)grow * N + gcol] = vv;
        if (Ch)   Ch[(size_t)grow * N + gcol] = (_Float16)vv;
      }
}

// ---------------------------------------------------------------------------
// Flash-style causal attention with relative-position skew.
// One wave per 16-query tile; 4 waves / block (independent; no barriers).
// grid = B * H * (L/64), block = 128.
// rel[l,m] = q[l] . Er[L-1+m-l]  (valid for causal m<=l)
// ---------------------------------------------------------------------------
__global__ __launch_bounds__(128) void attn_kernel(
    const _Float16* __restrict__ q, const _Float16* __restrict__ k,
    const _Float16* __restrict__ v, const _Float16* __restrict__ er,
    const float* __restrict__ npd, _Float16* __restrict__ out16) {
  const int L = 2048, D = 512, DK = 64;
  __shared__ _Float16 Pl[4][16 * 32];  // per-wave probability tile (16x32, f16)

  const int lane = threadIdx.x & 31, wid = threadIdx.x >> 5;
  const int row16 = lane & 15, hi = lane >> 4, koff = hi << 4;
  int blk = blockIdx.x;                // [b][h][L/64]
  int l0 = (blk & 31) * 64 + wid * 16; // L/64 == 32
  int hh = (blk >> 5) & 7;
  int b  = blk >> 8;

  const _Float16* qb = q + ((size_t)b * L) * D + hh * DK;
  const _Float16* kb = k + ((size_t)b * L) * D + hh * DK;
  const _Float16* vb = v + ((size_t)b * L) * D + hh * DK;
  const _Float16* eb = er + (size_t)hh * L * DK;
  const float*   npb = npd + (size_t)b * L;

  // Persistent Q fragments (16 rows x K=64 -> two 16x32 A-frags)
  v16h qa[2];
  {
    const _Float16* p = qb + (size_t)(l0 + row16) * D;
    qa[0] = pack2(p + koff,      p + 8 + koff);
    qa[1] = pack2(p + 32 + koff, p + 40 + koff);
  }

  float mrow[8], lrow[8];
  v8f oacc[4] = {};
#pragma unroll
  for (int r = 0; r < 8; ++r) { mrow[r] = -1e30f; lrow[r] = 0.0f; }
  const float scale = 0.125f;          // 1/sqrt(64)
  const int lmax = l0 + 15;

  for (int m0g = 0; m0g <= lmax; m0g += 32) {
    // prefetch next key/value tiles into cache while this tile computes
    if (m0g + 32 <= lmax) {
      __builtin_prefetch(kb + (size_t)(m0g + 32 + row16) * D, 0, 1);
      __builtin_prefetch(vb + (size_t)(m0g + 32 + row16) * D, 0, 1);
    }
    // hoist V fragments: independent of the score/softmax dependency chain
    v16h vf[4];
#pragma unroll
    for (int nt = 0; nt < 4; ++nt)
#pragma unroll
      for (int i = 0; i < 16; ++i) {
        int m = m0g + (hi << 4) + i;
        vf[nt][i] = vb[(size_t)m * D + nt * 16 + row16];
      }

    v8f S[2];
#pragma unroll
    for (int st = 0; st < 2; ++st) {
      const int m0 = m0g + st * 16;
      v8f sacc = {};
      if (m0 <= lmax) {
        v8f p0 = {}, p1 = {};
        const int base = L - 1 - l0 + m0;
#pragma unroll
        for (int kc = 0; kc < 2; ++kc) {
          // K^T fragment: col = key (m0+row16), K-dim = head dim (contiguous)
          const _Float16* kp = kb + (size_t)(m0 + row16) * D + kc * 32 + (hi << 4);
          v16h kf = pack2(kp, kp + 8);
          sacc = WMMA16(qa[kc], kf, sacc);
          // Er band fragments: rows base..base+15 (P1) and base-16..base-1 (P0)
          v16h ef = {};
          int e1 = base + row16;
          if ((unsigned)e1 < (unsigned)L) {
            const _Float16* ep = eb + (size_t)e1 * DK + kc * 32 + (hi << 4);
            ef = pack2(ep, ep + 8);
          }
          p1 = WMMA16(qa[kc], ef, p1);
          v16h ef0 = {};
          int e0 = base - 16 + row16;
          if ((unsigned)e0 < (unsigned)L) {
            const _Float16* ep = eb + (size_t)e0 * DK + kc * 32 + (hi << 4);
            ef0 = pack2(ep, ep + 8);
          }
          p0 = WMMA16(qa[kc], ef0, p0);
        }
        // gather diagonal rel entries via cross-lane shuffle, scale + mask
#pragma unroll
        for (int r = 0; r < 8; ++r) {
          int i = r + (hi << 3), j = row16;
          int d = j - i;
          int srcl = ((d >= 0) ? d : (16 + d)) + (hi << 4);
          float r0 = __shfl(p0[r], srcl, 32);
          float r1 = __shfl(p1[r], srcl, 32);
          float rel = (d >= 0) ? r1 : r0;
          int m = m0 + j;
          bool ok = (m <= l0 + i) && (npb[m] != 0.0f);
          sacc[r] = ok ? (sacc[r] + rel) * scale : -1e9f;
        }
      } else {
#pragma unroll
        for (int r = 0; r < 8; ++r) sacc[r] = -1e9f;
      }
      S[st] = sacc;
    }

    // online softmax (rows live in 16-lane halves -> xor-reduce masks 8..1)
    float alpha[8];
#pragma unroll
    for (int r = 0; r < 8; ++r) {
      float vmax = fmaxf(S[0][r], S[1][r]);
#pragma unroll
      for (int o = 8; o > 0; o >>= 1) vmax = fmaxf(vmax, __shfl_xor(vmax, o, 32));
      float nm  = fmaxf(mrow[r], vmax);
      float a   = __expf(mrow[r] - nm);
      float p0e = __expf(S[0][r] - nm);
      float p1e = __expf(S[1][r] - nm);
      float rs  = p0e + p1e;
#pragma unroll
      for (int o = 8; o > 0; o >>= 1) rs += __shfl_xor(rs, o, 32);
      mrow[r] = nm;
      lrow[r] = lrow[r] * a + rs;
      alpha[r] = a;
      S[0][r] = p0e; S[1][r] = p1e;
    }
#pragma unroll
    for (int nt = 0; nt < 4; ++nt)
#pragma unroll
      for (int r = 0; r < 8; ++r) oacc[nt][r] *= alpha[r];

    // C-layout fp32 P -> f16 A-fragment via LDS round trip (same-wave RAW)
#pragma unroll
    for (int st = 0; st < 2; ++st)
#pragma unroll
      for (int r = 0; r < 8; ++r)
        Pl[wid][(r + (hi << 3)) * 32 + st * 16 + row16] = (_Float16)S[st][r];
    asm volatile("s_wait_dscnt 0" ::: "memory");
    v16h pa;
    {
      const _Float16* p = &Pl[wid][row16 * 32 + koff];
      pa = pack2(p, p + 8);
    }
    // P(16x32) @ V(32x64): four 16-col tiles of V as B-fragments
#pragma unroll
    for (int nt = 0; nt < 4; ++nt) oacc[nt] = WMMA16(pa, vf[nt], oacc[nt]);
  }

  // normalize and store f16 output
#pragma unroll
  for (int nt = 0; nt < 4; ++nt)
#pragma unroll
    for (int r = 0; r < 8; ++r) {
      int i = r + (hi << 3);
      float val = oacc[nt][r] / lrow[r];
      out16[((size_t)b * L + l0 + i) * D + hh * DK + nt * 16 + row16] = (_Float16)val;
    }
}

// ---------------------------------------------------------------------------
// LayerNorm over D=512 + non-pad mask; one wave per row (16 elems / lane).
// Writes fp32 result and f16 mirror. grid = (B*L)/8, block = 256.
// ---------------------------------------------------------------------------
__global__ __launch_bounds__(256) void ln_kernel(
    const float* __restrict__ in, const float* __restrict__ g,
    const float* __restrict__ bb, const float* __restrict__ npd,
    float* __restrict__ hout, _Float16* __restrict__ h16) {
  const int D = 512;
  const int lane = threadIdx.x & 31, wid = threadIdx.x >> 5;
  size_t row = (size_t)blockIdx.x * 8 + wid;
  const float* p = in + row * D;
  float vals[16];
  float s = 0.0f;
#pragma unroll
  for (int i = 0; i < 16; ++i) { vals[i] = p[lane + i * 32]; s += vals[i]; }
#pragma unroll
  for (int o = 16; o > 0; o >>= 1) s += __shfl_xor(s, o, 32);
  float mu = s * (1.0f / 512.0f);
  float vs = 0.0f;
#pragma unroll
  for (int i = 0; i < 16; ++i) { float dd = vals[i] - mu; vs += dd * dd; }
#pragma unroll
  for (int o = 16; o > 0; o >>= 1) vs += __shfl_xor(vs, o, 32);
  float inv = rsqrtf(vs * (1.0f / 512.0f) + 1e-6f);
  float np = npd[row];
#pragma unroll
  for (int i = 0; i < 16; ++i) {
    int c = lane + i * 32;
    float ov = (g[c] * (vals[i] - mu) * inv + bb[c]) * np;
    hout[row * D + c] = ov;
    h16[row * D + c]  = (_Float16)ov;
  }
}

// ---------------------------------------------------------------------------
// Host orchestration
// ---------------------------------------------------------------------------
extern "C" void kernel_launch(void* const* d_in, const int* in_sizes, int n_in,
                              void* d_out, int out_size, void* d_ws,
                              size_t ws_size, hipStream_t stream) {
  (void)in_sizes; (void)n_in; (void)out_size; (void)ws_size;
  const float* x   = (const float*)d_in[0];
  const float* pos = (const float*)d_in[1];
  const float* Wq  = (const float*)d_in[2];
  const float* Wk  = (const float*)d_in[3];
  const float* Wv  = (const float*)d_in[4];
  const float* Er  = (const float*)d_in[5];
  const float* Wo  = (const float*)d_in[6];
  const float* g1  = (const float*)d_in[7];
  const float* b1  = (const float*)d_in[8];
  const float* W1  = (const float*)d_in[9];
  const float* W2  = (const float*)d_in[10];
  const float* g2  = (const float*)d_in[11];
  const float* b2  = (const float*)d_in[12];

  constexpr int B = 2, L = 2048, D = 512, DI = 2048, H = 8, NL = 4;
  constexpr int M = B * L;

  size_t off = 0;
  char* wsb = (char*)d_ws;
  auto wsa = [&](size_t bytes) -> char* {
    char* p = wsb + off;
    off += (bytes + 255) & ~(size_t)255;
    return p;
  };
  float*    h     = (float*)wsa((size_t)M * D * 4);
  _Float16* h16   = (_Float16*)wsa((size_t)M * D * 2);
  _Float16* q16   = (_Float16*)wsa((size_t)M * D * 2);
  _Float16* k16   = (_Float16*)wsa((size_t)M * D * 2);
  _Float16* v16v  = (_Float16*)wsa((size_t)M * D * 2);
  _Float16* o16   = (_Float16*)wsa((size_t)M * D * 2);
  float*    tmp   = (float*)wsa((size_t)M * D * 4);
  _Float16* mid16 = (_Float16*)wsa((size_t)M * DI * 2);
  _Float16* wq16  = (_Float16*)wsa((size_t)NL * D * D * 2);
  _Float16* wk16  = (_Float16*)wsa((size_t)NL * D * D * 2);
  _Float16* wv16  = (_Float16*)wsa((size_t)NL * D * D * 2);
  _Float16* wo16  = (_Float16*)wsa((size_t)NL * D * D * 2);
  _Float16* w116  = (_Float16*)wsa((size_t)NL * D * DI * 2);
  _Float16* w216  = (_Float16*)wsa((size_t)NL * DI * D * 2);
  _Float16* er16  = (_Float16*)wsa((size_t)NL * H * L * 64 * 2);
  float*    npd   = (float*)wsa((size_t)M * 4);

  auto cvt = [&](const float* s, _Float16* dptr, int n) {
    int grid = (n + 255) / 256;
    if (grid > 4096) grid = 4096;
    cvt_f32_to_f16<<<grid, 256, 0, stream>>>(s, dptr, n);
  };
  cvt(Wq, wq16, NL * D * D);
  cvt(Wk, wk16, NL * D * D);
  cvt(Wv, wv16, NL * D * D);
  cvt(Wo, wo16, NL * D * D);
  cvt(W1, w116, NL * D * DI);
  cvt(W2, w216, NL * DI * D);
  cvt(Er, er16, NL * H * L * 64);

  prep_kernel<<<M, 256, 0, stream>>>(x, pos, h, h16, npd);

  for (int ly = 0; ly < NL; ++ly) {
    const size_t wo = (size_t)ly * D * D;
    gemm_f16_kernel<<<dim3(D / 128, M / 128), 256, 0, stream>>>(
        h16, wq16 + wo, M, D, D, nullptr, nullptr, q16, 0);
    gemm_f16_kernel<<<dim3(D / 128, M / 128), 256, 0, stream>>>(
        h16, wk16 + wo, M, D, D, nullptr, nullptr, k16, 0);
    gemm_f16_kernel<<<dim3(D / 128, M / 128), 256, 0, stream>>>(
        h16, wv16 + wo, M, D, D, nullptr, nullptr, v16v, 0);

    attn_kernel<<<dim3(B * H * (L / 64)), 128, 0, stream>>>(
        q16, k16, v16v, er16 + (size_t)ly * H * L * 64, npd, o16);

    gemm_f16_kernel<<<dim3(D / 128, M / 128), 256, 0, stream>>>(
        o16, wo16 + wo, M, D, D, h, tmp, nullptr, 0);
    ln_kernel<<<M / 8, 256, 0, stream>>>(tmp, g1 + ly * D, b1 + ly * D, npd, h, h16);

    gemm_f16_kernel<<<dim3(DI / 128, M / 128), 256, 0, stream>>>(
        h16, w116 + (size_t)ly * D * DI, M, DI, D, nullptr, nullptr, mid16, 1);
    gemm_f16_kernel<<<dim3(D / 128, M / 128), 256, 0, stream>>>(
        mid16, w216 + (size_t)ly * DI * D, M, D, DI, h, tmp, nullptr, 0);

    float* outp = (ly == NL - 1) ? (float*)d_out : h;
    ln_kernel<<<M / 8, 256, 0, stream>>>(tmp, g2 + ly * D, b2 + ly * D, npd, outp, h16);
  }
}
